// TRNN_19542101197481
// MI455X (gfx1250) — matrix operand order repaired and checked by reading, not compiled, not run on hardware
//
#include <hip/hip_runtime.h>
#include <hip/hip_bf16.h>
#include <stdint.h>
#include <stddef.h>

// ---------------------------------------------------------------------------
// 2-layer GRU for MI455X (gfx1250, wave32).
// bf16 WMMA 16x16x32 (f32 accum); weights pre-tiled bf16, L2-resident.
// Recurrent operand panels ([h] / [z*h]) staged into LDS by the Tensor Data
// Mover (tensor_load_to_lds, double-buffered, s_wait_tensorcnt), consumed via
// conflict-free ds_load_b128 (padded 36-dword LDS row stride).
// Persistent 16-WG kernel, two device-scope barriers per time step.
// ---------------------------------------------------------------------------

static constexpr int kB       = 128;   // batch
static constexpr int kT       = 512;   // sequence length
static constexpr int kDout    = 512;   // hidden dim
static constexpr int kDin0    = 256;   // layer-0 input dim
static constexpr int kNWG     = 16;    // persistent workgroups
static constexpr int kThreads = 256;   // 8 waves of 32

static constexpr int kPanelCols  = 64;              // K-columns per LDS panel
static constexpr int kNumPanels  = kDout / kPanelCols;   // 8
static constexpr int kLdsStride  = kPanelCols + 8;  // 72 bf16 = 36 dwords/row

typedef __hip_bfloat16 bf16;
typedef __attribute__((ext_vector_type(16))) __bf16 v16bf;
typedef __attribute__((ext_vector_type(8)))  float  v8f;
typedef __attribute__((ext_vector_type(4)))  uint32_t v4u;
typedef __attribute__((ext_vector_type(8)))  int      v8i;
typedef __attribute__((ext_vector_type(4)))  int      v4i;

struct __align__(16) U4 { uint32_t x, y, z, w; };
union FragU { U4 q[2]; v16bf v; };

#if __has_builtin(__builtin_amdgcn_tensor_load_to_lds) && \
    __has_builtin(__builtin_amdgcn_s_wait_tensorcnt)
#define GRU_USE_TDM 1
#else
#define GRU_USE_TDM 0
#endif

// ---------------------------------------------------------------------------
// Device-scope sense-reversing grid barrier (spin with s_sleep).
// ---------------------------------------------------------------------------
__device__ __forceinline__ void grid_barrier(unsigned* cnt, unsigned* gen,
                                             unsigned nwg) {
  __syncthreads();
  if (threadIdx.x == 0) {
    __threadfence();
    unsigned g = __hip_atomic_load(gen, __ATOMIC_RELAXED,
                                   __HIP_MEMORY_SCOPE_AGENT);
    if (__hip_atomic_fetch_add(cnt, 1u, __ATOMIC_ACQ_REL,
                               __HIP_MEMORY_SCOPE_AGENT) == nwg - 1u) {
      __hip_atomic_store(cnt, 0u, __ATOMIC_RELAXED, __HIP_MEMORY_SCOPE_AGENT);
      __hip_atomic_fetch_add(gen, 1u, __ATOMIC_ACQ_REL,
                             __HIP_MEMORY_SCOPE_AGENT);
    } else {
      while (__hip_atomic_load(gen, __ATOMIC_ACQUIRE,
                               __HIP_MEMORY_SCOPE_AGENT) == g) {
        __builtin_amdgcn_s_sleep(1);
      }
    }
  }
  __syncthreads();
}

// ---------------------------------------------------------------------------
// TDM: DMA one 64x128 bf16 K-panel (rows = batch 128, cols = 64 of 512) from
// global [128 x 512] into LDS with a 36-dword padded row stride.
// Issued by wave 0 only; completion via TENSORcnt.
// ---------------------------------------------------------------------------
__device__ __forceinline__ void panel_issue(bf16* lds_dst, const bf16* gsrc,
                                            int wave) {
#if GRU_USE_TDM
  if (wave == 0) {
    const uint64_t ga = (uint64_t)(uintptr_t)gsrc;       // global byte addr
    const uint32_t la = (uint32_t)(uintptr_t)lds_dst;    // LDS byte offset
    v4u g0;
    g0[0] = 1u;                                   // count=1, user descriptor
    g0[1] = la;                                   // lds_addr
    g0[2] = (uint32_t)ga;                         // global_addr[31:0]
    g0[3] = (uint32_t)((ga >> 32) & 0x1ffffffu)   // global_addr[56:32]
            | (2u << 30);                         // type = 2 ("image")
    v8i g1;
    // data_size=2B (code 1) | pad_enable | pad_interval=32dw (code 4)
    // | pad_amount=4dw (code 3)
    g1[0] = (int)((1u << 16) | (1u << 20) | (4u << 22) | (3u << 25));
    g1[1] = (int)((uint32_t)kDout << 16);         // tensor_dim0 = 512
    g1[2] = (int)((uint32_t)kB << 16);            // tensor_dim1 = 128
    g1[3] = (int)((uint32_t)kPanelCols << 16);    // tile_dim0 = 64
    g1[4] = kB;                                   // tile_dim1 = 128
    g1[5] = kDout;                                // tensor_dim0_stride = 512
    g1[6] = 0;
    g1[7] = 0;
    v4i z4 = {0, 0, 0, 0};
    v8i z8 = {0, 0, 0, 0, 0, 0, 0, 0};
    __builtin_amdgcn_tensor_load_to_lds(g0, g1, z4, z4, z8, 0);
  }
#else
  // Fallback: cooperative copy through VGPRs into padded LDS layout.
  for (int i = (int)threadIdx.x; i < kB * kPanelCols; i += kThreads) {
    const int r = i >> 6;
    const int c = i & (kPanelCols - 1);
    lds_dst[r * kLdsStride + c] = gsrc[(size_t)r * kDout + c];
  }
#endif
}

__device__ __forceinline__ void panel_wait(bool more_inflight) {
#if GRU_USE_TDM
  if (more_inflight) {
    __builtin_amdgcn_s_wait_tensorcnt(1);   // allow next panel DMA in flight
  } else {
    __builtin_amdgcn_s_wait_tensorcnt(0);   // drain
  }
#endif
  __syncthreads();
}

// ---------------------------------------------------------------------------
// x [B,T,Din] f32  ->  [T,B,Din] bf16
// ---------------------------------------------------------------------------
__global__ void gru_pack_x(const float* __restrict__ src,
                           bf16* __restrict__ dst) {
  size_t i = (size_t)blockIdx.x * blockDim.x + threadIdx.x;
  const size_t total = (size_t)kT * kB * kDin0;
  if (i >= total) return;
  int d = (int)(i % kDin0);
  size_t r = i / kDin0;
  int b = (int)(r % kB);
  int t = (int)(r / kB);
  dst[i] = __float2bfloat16(src[((size_t)b * kT + t) * kDin0 + d]);
}

// ---------------------------------------------------------------------------
// W [K,N] row-major f32 -> bf16 tiled for WMMA B fragments:
//   ((ks*NB + nb)*32 + lane)*16 + j = W[ks*32 + (lane>=16?16:0)+j,
//                                       nb*16 + (lane&15)]
// ---------------------------------------------------------------------------
__global__ void gru_pack_w(const float* __restrict__ src,
                           bf16* __restrict__ dst, int K, int N) {
  int idx = blockIdx.x * blockDim.x + threadIdx.x;
  const int NB = N >> 4;
  const int total = (K >> 5) * NB * 32;
  if (idx >= total) return;
  const int lane = idx & 31;
  const int tile = idx >> 5;
  const int nb = tile % NB;
  const int ks = tile / NB;
  const int n  = (nb << 4) + (lane & 15);
  const int k0 = (ks << 5) + ((lane >> 4) << 4);
  bf16* d = dst + (size_t)idx * 16;
#pragma unroll
  for (int j = 0; j < 16; ++j)
    d[j] = __float2bfloat16(src[(size_t)(k0 + j) * N + n]);
}

// ---------------------------------------------------------------------------
// One GRU layer, persistent kernel over all T steps.
// WG g owns hidden columns [g*32, g*32+32).
// Wave (a = wave>>1, wb = wave&1): rows 32a..32a+31; wb selects z vs r half.
// ---------------------------------------------------------------------------
__global__ __launch_bounds__(kThreads) void gru_layer_kernel(
    const bf16* __restrict__ xseq, int Din,
    const bf16* __restrict__ Wzr, const bf16* __restrict__ Wh,
    const float* __restrict__ b_zr, const float* __restrict__ b_h,
    bf16* __restrict__ h_buf, bf16* __restrict__ zh_buf,
    float* __restrict__ r_buf,
    bf16* __restrict__ out_bf, float* __restrict__ out_f32,
    unsigned* bar_cnt, unsigned* bar_gen) {
  __shared__ __align__(16) bf16 s_panel[2][kB * kLdsStride];

  const int wave = (int)(threadIdx.x >> 5);
  const int lane = (int)(threadIdx.x & 31);
  const int a  = wave >> 1;                 // row quad: rows 32a .. 32a+31
  const int wb = wave & 1;                  // 0 => z half, 1 => r half
  const int colbase = (int)blockIdx.x * 32; // owned hidden columns
  const int ksx = Din >> 5;                 // x-region K-steps
  const int NBzr = (2 * kDout) >> 4;        // 64 column tiles in zr
  const int NBh  = kDout >> 4;              // 32 column tiles in h
  const int lo   = lane & 15;
  const int hiK  = (lane >> 4) << 4;        // A-frag K sub-offset (0 or 16)
  const int hiM  = (lane >> 4) << 3;        // C-frag row sub-offset (0 or 8)

  // Warm L2 with this layer's bf16 weights (global_prefetch_b8).
  {
    const int K = Din + kDout;
    const char* w0 = (const char*)Wzr;
    const char* w1 = (const char*)Wh;
    const size_t s0 = (size_t)K * (2 * kDout) * 2;
    const size_t s1 = (size_t)K * kDout * 2;
    const size_t stride = (size_t)kNWG * kThreads * 256;
    const size_t base = ((size_t)blockIdx.x * kThreads + threadIdx.x) * 256;
    for (size_t o = base; o < s0; o += stride) __builtin_prefetch(w0 + o, 0, 1);
    for (size_t o = base; o < s1; o += stride) __builtin_prefetch(w1 + o, 0, 1);
  }

  for (int t = 0; t < kT; ++t) {
    const bf16* xt = xseq + (size_t)t * kB * Din;

    // ==== Phase 1: zr = sigmoid([x_t | h] @ W_zr + b_zr) ==================
    v8f acc[2][2] = {};

    // --- x-region: A fragments straight from the L2-resident x panel ---
    for (int ks = 0; ks < ksx; ++ks) {
      const int kb = (ks << 5) + hiK;
      v16bf afr[2];
#pragma unroll
      for (int rt = 0; rt < 2; ++rt) {
        const int row = ((a * 2 + rt) << 4) + lo;
        const bf16* p = xt + (size_t)row * Din + kb;
        FragU f;
        f.q[0] = *(const U4*)p;
        f.q[1] = *(const U4*)(p + 8);
        afr[rt] = f.v;
      }
#pragma unroll
      for (int ct = 0; ct < 2; ++ct) {
        const int nb = (wb ? (NBzr >> 1) : 0) + (colbase >> 4) + ct;
        const bf16* bp = Wzr + ((((size_t)ks * NBzr + nb) << 5) + lane) * 16;
        FragU fb;
        fb.q[0] = *(const U4*)bp;
        fb.q[1] = *(const U4*)(bp + 8);
        const v16bf bfr = fb.v;
#pragma unroll
        for (int rt = 0; rt < 2; ++rt)
          acc[rt][ct] = __builtin_amdgcn_wmma_f32_16x16x32_bf16(
              false, afr[rt], false, bfr, (short)0, acc[rt][ct], false, false);
      }
    }

    // --- recurrent region: h panels DMA'd into LDS, double-buffered ---
    panel_issue(&s_panel[0][0], h_buf, wave);
    for (int p = 0; p < kNumPanels; ++p) {
      if (p + 1 < kNumPanels)
        panel_issue(&s_panel[(p + 1) & 1][0], h_buf + (p + 1) * kPanelCols,
                    wave);
      panel_wait(p + 1 < kNumPanels);
      const bf16* lpb = &s_panel[p & 1][0];
#pragma unroll
      for (int ksl = 0; ksl < 2; ++ksl) {
        const int ks = ksx + p * 2 + ksl;
        const int lkb = (ksl << 5) + hiK;
        v16bf afr[2];
#pragma unroll
        for (int rt = 0; rt < 2; ++rt) {
          const int row = ((a * 2 + rt) << 4) + lo;
          const bf16* lp = lpb + row * kLdsStride + lkb;
          FragU f;                       // ds_load_b128 x2, conflict-free
          f.q[0] = *(const U4*)lp;
          f.q[1] = *(const U4*)(lp + 8);
          afr[rt] = f.v;
        }
#pragma unroll
        for (int ct = 0; ct < 2; ++ct) {
          const int nb = (wb ? (NBzr >> 1) : 0) + (colbase >> 4) + ct;
          const bf16* bp = Wzr + ((((size_t)ks * NBzr + nb) << 5) + lane) * 16;
          FragU fb;
          fb.q[0] = *(const U4*)bp;
          fb.q[1] = *(const U4*)(bp + 8);
          const v16bf bfr = fb.v;
#pragma unroll
          for (int rt = 0; rt < 2; ++rt)
            acc[rt][ct] = __builtin_amdgcn_wmma_f32_16x16x32_bf16(
                false, afr[rt], false, bfr, (short)0, acc[rt][ct], false,
                false);
        }
      }
      __syncthreads();
    }

    // --- sigmoid; stage z*h (bf16) and r (f32) ---
#pragma unroll
    for (int rt = 0; rt < 2; ++rt) {
#pragma unroll
      for (int ct = 0; ct < 2; ++ct) {
        const int colh = colbase + (ct << 4) + lo;  // hidden column 0..511
        const float bias = b_zr[colh + (wb ? kDout : 0)];
#pragma unroll
        for (int j = 0; j < 8; ++j) {
          const int row = ((a * 2 + rt) << 4) + hiM + j;
          const float s = 1.0f / (1.0f + __expf(-(acc[rt][ct][j] + bias)));
          if (wb == 0) {
            const float hp =
                __bfloat162float(h_buf[(size_t)row * kDout + colh]);
            zh_buf[(size_t)row * kDout + colh] = __float2bfloat16(s * hp);
          } else {
            r_buf[(size_t)row * kDout + colh] = s;
          }
        }
      }
    }

    grid_barrier(bar_cnt, bar_gen, kNWG);

    // ==== Phase 2: h~ = tanh([x_t | z*h] @ W_h + b_h); h = r*h+(1-r)*h~ ===
    v8f acc2[2] = {};
    const int nb2 = (colbase >> 4) + wb;

    for (int ks = 0; ks < ksx; ++ks) {
      const int kb = (ks << 5) + hiK;
      const bf16* bp = Wh + ((((size_t)ks * NBh + nb2) << 5) + lane) * 16;
      FragU fb;
      fb.q[0] = *(const U4*)bp;
      fb.q[1] = *(const U4*)(bp + 8);
      const v16bf bfr = fb.v;
#pragma unroll
      for (int rt = 0; rt < 2; ++rt) {
        const int row = ((a * 2 + rt) << 4) + lo;
        const bf16* p = xt + (size_t)row * Din + kb;
        FragU f;
        f.q[0] = *(const U4*)p;
        f.q[1] = *(const U4*)(p + 8);
        acc2[rt] = __builtin_amdgcn_wmma_f32_16x16x32_bf16(
            false, f.v, false, bfr, (short)0, acc2[rt], false, false);
      }
    }

    panel_issue(&s_panel[0][0], zh_buf, wave);
    for (int p = 0; p < kNumPanels; ++p) {
      if (p + 1 < kNumPanels)
        panel_issue(&s_panel[(p + 1) & 1][0], zh_buf + (p + 1) * kPanelCols,
                    wave);
      panel_wait(p + 1 < kNumPanels);
      const bf16* lpb = &s_panel[p & 1][0];
#pragma unroll
      for (int ksl = 0; ksl < 2; ++ksl) {
        const int ks = ksx + p * 2 + ksl;
        const int lkb = (ksl << 5) + hiK;
        const bf16* bp = Wh + ((((size_t)ks * NBh + nb2) << 5) + lane) * 16;
        FragU fb;
        fb.q[0] = *(const U4*)bp;
        fb.q[1] = *(const U4*)(bp + 8);
        const v16bf bfr = fb.v;
#pragma unroll
        for (int rt = 0; rt < 2; ++rt) {
          const int row = ((a * 2 + rt) << 4) + lo;
          const bf16* lp = lpb + row * kLdsStride + lkb;
          FragU f;
          f.q[0] = *(const U4*)lp;
          f.q[1] = *(const U4*)(lp + 8);
          acc2[rt] = __builtin_amdgcn_wmma_f32_16x16x32_bf16(
              false, f.v, false, bfr, (short)0, acc2[rt], false, false);
        }
      }
      __syncthreads();
    }

    {
      const int colh = colbase + (wb << 4) + lo;
      const float bias = b_h[colh];
#pragma unroll
      for (int rt = 0; rt < 2; ++rt) {
#pragma unroll
        for (int j = 0; j < 8; ++j) {
          const int row = ((a * 2 + rt) << 4) + hiM + j;
          const float v = acc2[rt][j] + bias;
          const float e = __expf(-2.0f * v);       // tanh via exp
          const float ht = (1.0f - e) / (1.0f + e);
          const float r  = r_buf[(size_t)row * kDout + colh];
          const float hp =
              __bfloat162float(h_buf[(size_t)row * kDout + colh]);
          const float hn = r * hp + (1.0f - r) * ht;
          h_buf[(size_t)row * kDout + colh] = __float2bfloat16(hn);
          if (out_f32) {  // final layer: f32 output [B,T,Dout]
            out_f32[((size_t)row * kT + t) * kDout + colh] = hn;
          } else {        // inner layer: bf16 sequence [T,B,Dout]
            out_bf[((size_t)t * kB + row) * kDout + colh] =
                __float2bfloat16(hn);
          }
        }
      }
    }

    grid_barrier(bar_cnt, bar_gen, kNWG);
  }
}

// ---------------------------------------------------------------------------
// Workspace layout (bytes)
// ---------------------------------------------------------------------------
static constexpr size_t OFF_CNT  = 0;
static constexpr size_t OFF_H    = 256;
static constexpr size_t OFF_ZH   = OFF_H   + (size_t)kB * kDout * 2;
static constexpr size_t OFF_R    = OFF_ZH  + (size_t)kB * kDout * 2;
static constexpr size_t OFF_X0   = OFF_R   + (size_t)kB * kDout * 4;
static constexpr size_t OFF_H0S  = OFF_X0  + (size_t)kT * kB * kDin0 * 2;
static constexpr size_t OFF_WZR0 = OFF_H0S + (size_t)kT * kB * kDout * 2;
static constexpr size_t OFF_WH0  = OFF_WZR0 + (size_t)(kDin0 + kDout) * (2 * kDout) * 2;
static constexpr size_t OFF_WZR1 = OFF_WH0  + (size_t)(kDin0 + kDout) * kDout * 2;
static constexpr size_t OFF_WH1  = OFF_WZR1 + (size_t)(2 * kDout) * (2 * kDout) * 2;

extern "C" void kernel_launch(void* const* d_in, const int* in_sizes, int n_in,
                              void* d_out, int out_size, void* d_ws,
                              size_t ws_size, hipStream_t stream) {
  (void)in_sizes; (void)n_in; (void)out_size; (void)ws_size;
  const float* x     = (const float*)d_in[0];
  const float* W_zr0 = (const float*)d_in[1];
  const float* b_zr0 = (const float*)d_in[2];
  const float* W_h0  = (const float*)d_in[3];
  const float* b_h0  = (const float*)d_in[4];
  const float* W_zr1 = (const float*)d_in[5];
  const float* b_zr1 = (const float*)d_in[6];
  const float* W_h1  = (const float*)d_in[7];
  const float* b_h1  = (const float*)d_in[8];
  float* out = (float*)d_out;
  char* ws = (char*)d_ws;

  unsigned* bar_cnt = (unsigned*)(ws + OFF_CNT);
  unsigned* bar_gen = bar_cnt + 1;
  bf16* h_buf  = (bf16*)(ws + OFF_H);
  bf16* zh_buf = (bf16*)(ws + OFF_ZH);
  float* r_buf = (float*)(ws + OFF_R);
  bf16* x0seq  = (bf16*)(ws + OFF_X0);
  bf16* h0seq  = (bf16*)(ws + OFF_H0S);
  bf16* wzr0   = (bf16*)(ws + OFF_WZR0);
  bf16* wh0    = (bf16*)(ws + OFF_WH0);
  bf16* wzr1   = (bf16*)(ws + OFF_WZR1);
  bf16* wh1    = (bf16*)(ws + OFF_WH1);

  // Zero barrier state + h + staging buffers.
  (void)hipMemsetAsync(ws, 0, OFF_X0, stream);

  // Pack x and all four weight matrices to bf16 WMMA layouts.
  {
    size_t total = (size_t)kT * kB * kDin0;
    gru_pack_x<<<dim3((unsigned)((total + 255) / 256)), dim3(256), 0, stream>>>(
        x, x0seq);
  }
  {
    int tot = ((kDin0 + kDout) / 32) * ((2 * kDout) / 16) * 32;
    gru_pack_w<<<dim3((tot + 255) / 256), dim3(256), 0, stream>>>(
        W_zr0, wzr0, kDin0 + kDout, 2 * kDout);
  }
  {
    int tot = ((kDin0 + kDout) / 32) * (kDout / 16) * 32;
    gru_pack_w<<<dim3((tot + 255) / 256), dim3(256), 0, stream>>>(
        W_h0, wh0, kDin0 + kDout, kDout);
  }
  {
    int tot = ((2 * kDout) / 32) * ((2 * kDout) / 16) * 32;
    gru_pack_w<<<dim3((tot + 255) / 256), dim3(256), 0, stream>>>(
        W_zr1, wzr1, 2 * kDout, 2 * kDout);
  }
  {
    int tot = ((2 * kDout) / 32) * (kDout / 16) * 32;
    gru_pack_w<<<dim3((tot + 255) / 256), dim3(256), 0, stream>>>(
        W_h1, wh1, 2 * kDout, kDout);
  }

  // Layer 0: x0seq (bf16) -> h0seq (bf16).
  gru_layer_kernel<<<dim3(kNWG), dim3(kThreads), 0, stream>>>(
      x0seq, kDin0, wzr0, wh0, b_zr0, b_h0, h_buf, zh_buf, r_buf, h0seq,
      nullptr, bar_cnt, bar_gen);

  // Reset h and barrier state for layer 1.
  (void)hipMemsetAsync(ws, 0, OFF_X0, stream);

  // Layer 1: h0seq (bf16) -> out (f32, [B,T,Dout]).
  gru_layer_kernel<<<dim3(kNWG), dim3(kThreads), 0, stream>>>(
      h0seq, kDout, wzr1, wh1, b_zr1, b_h1, h_buf, zh_buf, r_buf, nullptr,
      out, bar_cnt, bar_gen);
}
